// Attention_LoRA_FFT_9388798509069
// MI455X (gfx1250) — matrix operand order repaired and checked by reading, not compiled, not run on hardware
//
#include <hip/hip_runtime.h>

// ---------------------------------------------------------------------------
// CDNA5 (gfx1250) implementation of Attention + DCT-LoRA.
// GEMMs: v_wmma_f32_16x16x32_bf16 (wave32 WMMA, f32 accumulate).
// Flash attention: bf16 q/k/v packs, async global->LDS V staging,
// DPP row_ror softmax reductions (no ds_bpermute).
// ---------------------------------------------------------------------------

typedef __attribute__((ext_vector_type(16))) __bf16 v16bf;
typedef __attribute__((ext_vector_type(8)))  float  v8f;
typedef __attribute__((ext_vector_type(4)))  int    v4i;

union FragAB { v16bf v; unsigned u[8]; };

__device__ __forceinline__ unsigned short bf16u(float a) {
    __bf16 h = (__bf16)a;
    return __builtin_bit_cast(unsigned short, h);
}
__device__ __forceinline__ float ubf(unsigned short u) {
    return (float)__builtin_bit_cast(__bf16, u);
}
__device__ __forceinline__ unsigned pk2(float a, float b) {
    return (unsigned)bf16u(a) | ((unsigned)bf16u(b) << 16);
}

// K index pattern for 16-bit A-operand fragments (CDNA5 ISA).
__device__ __forceinline__ int a_kk(int lane, int p) {
    int kh = (lane >> 4) * 8;
    return kh + ((p < 4) ? (2 * p) : (16 + 2 * (p - 4)));
}

// ---- DPP 16-lane ring reductions (VALU only, no LDS) ----------------------
#if __has_builtin(__builtin_amdgcn_update_dpp)
#define HAVE_DPP 1
template <int CTRL>
__device__ __forceinline__ float dpp_mov(float x) {
    return __builtin_bit_cast(float,
        __builtin_amdgcn_update_dpp(0, __builtin_bit_cast(int, x),
                                    CTRL, 0xF, 0xF, true));
}
__device__ __forceinline__ float red_max16(float x) {
    x = fmaxf(x, dpp_mov<0x128>(x));   // row_ror:8
    x = fmaxf(x, dpp_mov<0x124>(x));   // row_ror:4
    x = fmaxf(x, dpp_mov<0x122>(x));   // row_ror:2
    x = fmaxf(x, dpp_mov<0x121>(x));   // row_ror:1
    return x;
}
__device__ __forceinline__ float red_sum16(float x) {
    x += dpp_mov<0x128>(x);
    x += dpp_mov<0x124>(x);
    x += dpp_mov<0x122>(x);
    x += dpp_mov<0x121>(x);
    return x;
}
#else
__device__ __forceinline__ float red_max16(float x) {
    for (int off = 8; off >= 1; off >>= 1) x = fmaxf(x, __shfl_xor(x, off, 32));
    return x;
}
__device__ __forceinline__ float red_sum16(float x) {
    for (int off = 8; off >= 1; off >>= 1) x += __shfl_xor(x, off, 32);
    return x;
}
#endif

// ---- async global -> LDS (CDNA5 ASYNCcnt path) ----------------------------
// Signature (from hipcc diagnostic): arg0 = AS1 v4i* (global src),
// arg1 = AS3 v4i* (LDS dst), then imm offset + cpol.
#define GPTR(p) ((__attribute__((address_space(1))) v4i*)(p))
#define LPTR(p) ((__attribute__((address_space(3))) v4i*)(p))
#if __has_builtin(__builtin_amdgcn_global_load_async_to_lds_b128)
#define HAVE_ASYNC 1
#endif
__device__ __forceinline__ void wait_async0() {
#if __has_builtin(__builtin_amdgcn_s_wait_asynccnt)
    __builtin_amdgcn_s_wait_asynccnt(0);
#else
    asm volatile("s_wait_asynccnt 0x0" ::: "memory");
#endif
}

// ---------------------------------------------------------------------------
// DCT-II orthonormal matrix: Bm[i][j]
// ---------------------------------------------------------------------------
__global__ void build_dct_kernel(float* __restrict__ Bm, int n) {
    int idx = blockIdx.x * blockDim.x + threadIdx.x;
    if (idx >= n * n) return;
    int i = idx / n, j = idx % n;
    float val;
    if (i == 0) {
        val = rsqrtf((float)n);
    } else {
        int r = (i * (2 * j + 1)) % (4 * n);           // cos period reduction
        float ang = 3.14159265358979323846f * (float)r / (2.0f * (float)n);
        val = sqrtf(2.0f / (float)n) * __cosf(ang);
    }
    Bm[idx] = val;
}

__global__ void zero_f32_kernel(float* __restrict__ p, int n) {
    int i = blockIdx.x * blockDim.x + threadIdx.x;
    if (i < n) p[i] = 0.0f;
}

__global__ void scatter_coef_kernel(const float* __restrict__ coef,
                                    const int* __restrict__ indices,
                                    const int* __restrict__ taskp,
                                    float* __restrict__ S, int n_frq, int n_tasks) {
    int i = blockIdx.x * blockDim.x + threadIdx.x;
    int t = i / n_frq;
    if (t >= n_tasks || t > taskp[0]) return;
    atomicAdd(&S[indices[i]], coef[i]);
}

// ---------------------------------------------------------------------------
// Generic bf16-WMMA GEMM:  C(MxN) = op(A) @ op(B)   (f32 in)
// MODE: 0 = f32 store (+bias)
//       1 = qkv scatter into packed bf16 (B,H,N,64) q/k/v (q pre-scaled)
//       2 = bf16 read-add into packed (B,H,N,64) dst
// Block tile 64x128, BK=32, 256 threads (8 waves), wave tile 16x64 (4 accs).
// ---------------------------------------------------------------------------
#define GBM 64
#define GBN 128
#define GBK 32
#define ASTR 34    // A-tile LDS stride (bf16 elems), even -> dword aligned
#define BSTR 136   // B-tile LDS stride (128 + 8 pad)

template <int TRANSA, int TRANSB, int MODE>
__global__ void __launch_bounds__(256)
gemm_bf16_kernel(const float* __restrict__ A, int lda,
                 const float* __restrict__ B, int ldb,
                 float* __restrict__ C, int ldc,
                 const float* __restrict__ bias,
                 int M, int N, int K,
                 unsigned short* __restrict__ aux0,
                 unsigned short* __restrict__ aux1,
                 unsigned short* __restrict__ aux2) {
    __shared__ __align__(16) unsigned short As[GBM * ASTR]; // As[m][k]
    __shared__ __align__(16) unsigned short Bs[GBK * BSTR]; // Bs[k][n]

    const int tid  = threadIdx.x;
    const int lane = tid & 31;
    const int wave = tid >> 5;
    const int m0 = blockIdx.y * GBM;
    const int n0 = blockIdx.x * GBN;
    const int wm = (wave & 3) * 16;   // wave row offset inside block
    const int wn = (wave >> 2) * 64;  // wave col offset inside block

    v8f acc[4] = {v8f{}, v8f{}, v8f{}, v8f{}};

    for (int k0 = 0; k0 < K; k0 += GBK) {
        // ---- stage A tile 64x32 (512 float4 chunks, 2 per thread) ----
        if (TRANSA == 0) {
#pragma unroll
            for (int i = 0; i < 2; ++i) {
                int c = tid + i * 256;
                int r = c >> 3, c4 = (c & 7) * 4;
                const float4 f = *(const float4*)&A[(size_t)(m0 + r) * lda + k0 + c4];
                *(unsigned*)&As[r * ASTR + c4]     = pk2(f.x, f.y);
                *(unsigned*)&As[r * ASTR + c4 + 2] = pk2(f.z, f.w);
            }
        } else {
#pragma unroll
            for (int i = 0; i < 2; ++i) {
                int c = tid + i * 256;
                int ca = c >> 4, ra4 = (c & 15) * 4;   // coalesce over m
                const float4 f = *(const float4*)&A[(size_t)(k0 + ca) * lda + m0 + ra4];
                As[(ra4 + 0) * ASTR + ca] = bf16u(f.x);
                As[(ra4 + 1) * ASTR + ca] = bf16u(f.y);
                As[(ra4 + 2) * ASTR + ca] = bf16u(f.z);
                As[(ra4 + 3) * ASTR + ca] = bf16u(f.w);
            }
        }
        // ---- stage B tile 32x128 (1024 float4 chunks, 4 per thread) ----
        if (TRANSB == 0) {
#pragma unroll
            for (int i = 0; i < 4; ++i) {
                int c = tid + i * 256;
                int kb = c >> 5, nb4 = (c & 31) * 4;
                const float4 f = *(const float4*)&B[(size_t)(k0 + kb) * ldb + n0 + nb4];
                *(unsigned*)&Bs[kb * BSTR + nb4]     = pk2(f.x, f.y);
                *(unsigned*)&Bs[kb * BSTR + nb4 + 2] = pk2(f.z, f.w);
            }
        } else {
#pragma unroll
            for (int i = 0; i < 4; ++i) {
                int c = tid + i * 256;
                int kb4 = (c & 7) * 4, nb = c >> 3;    // coalesce over k
                const float4 f = *(const float4*)&B[(size_t)(n0 + nb) * ldb + k0 + kb4];
                Bs[(kb4 + 0) * BSTR + nb] = bf16u(f.x);
                Bs[(kb4 + 1) * BSTR + nb] = bf16u(f.y);
                Bs[(kb4 + 2) * BSTR + nb] = bf16u(f.z);
                Bs[(kb4 + 3) * BSTR + nb] = bf16u(f.w);
            }
        }
        __syncthreads();

        // ---- fragments + 4 WMMAs ----
        FragAB fa;
        const int rowA = wm + (lane & 15);
#pragma unroll
        for (int p = 0; p < 8; ++p)
            fa.u[p] = *(const unsigned*)&As[rowA * ASTR + a_kk(lane, p)];
#pragma unroll
        for (int j = 0; j < 4; ++j) {
            FragAB fb;   // B fragment: lane%16 = N column, K pairs per a_kk
#pragma unroll
            for (int p = 0; p < 8; ++p)
                fb.u[p] = (unsigned)Bs[a_kk(lane, p) * BSTR + wn + j * 16 + (lane & 15)] | ((unsigned)Bs[(a_kk(lane, p) + 1) * BSTR + wn + j * 16 + (lane & 15)] << 16);
            acc[j] = __builtin_amdgcn_wmma_f32_16x16x32_bf16(
                false, fa.v, false, fb.v, (short)0, acc[j], false, false);
        }
        __syncthreads();
    }

    // ---- epilogue: C layout VGPR r -> M = r + 8*(lane/16); N = lane%16 ----
    const int mrow = m0 + wm + (lane >> 4) * 8;
#pragma unroll
    for (int r = 0; r < 8; ++r) {
        int m = mrow + r;
        int b = m >> 10, ntok = m & 1023;
#pragma unroll
        for (int j = 0; j < 4; ++j) {
            int n = n0 + wn + j * 16 + (lane & 15);
            float val = acc[j][r];
            if (MODE == 0) {
                if (bias) val += bias[n];
                C[(size_t)m * ldc + n] = val;
            } else if (MODE == 1) {          // qkv split + head pack (bf16)
                int which = n >> 10, cc = n & 1023;
                int h = cc >> 6, d = cc & 63;
                unsigned short* base = (which == 0) ? aux0
                                     : (which == 1) ? aux1 : aux2;
                float w = (which == 0) ? val * 0.125f : val;   // fold softmax scale
                base[(((size_t)(b * 16 + h)) * 1024 + ntok) * 64 + d] = bf16u(w);
            } else {                         // MODE 2: LoRA add into bf16 pack
                int h = n >> 6, d = n & 63;
                size_t o = (((size_t)(b * 16 + h)) * 1024 + ntok) * 64 + d;
                aux0[o] = bf16u(ubf(aux0[o]) + val);
            }
        }
    }
}

// ---------------------------------------------------------------------------
// Flash attention: one block = 8 waves = 128 query rows of one (b,h).
// q,k,v packed bf16 (B*H, N, 64); q pre-scaled. Output (B, N, C) f32.
// ---------------------------------------------------------------------------
#define KTSTR 72   // K^T tile: [dim][token], 64 tokens + pad
#define VSTR  72   // V tile:   [token][dim], 64 dims + pad
#define PSTR  66   // per-wave P tile: [row][token]

__global__ void __launch_bounds__(256)
flash_attn_kernel(const unsigned short* __restrict__ Q,
                  const unsigned short* __restrict__ Kp,
                  const unsigned short* __restrict__ Vp,
                  float* __restrict__ O, int N) {
    __shared__ __align__(16) unsigned short KsT[64 * KTSTR];
    __shared__ __align__(16) unsigned short Vs[64 * VSTR];
    __shared__ __align__(16) unsigned short Ps[8 * 16 * PSTR];

    const int tid  = threadIdx.x;
    const int lane = tid & 31;
    const int wave = tid >> 5;
    const int bh = blockIdx.y;               // 0..127
    const int b = bh >> 4, h = bh & 15;
    const int r0 = blockIdx.x * 128 + wave * 16;

    const unsigned short* qb = Q + (size_t)bh * N * 64;
    const unsigned short* kb = Kp + (size_t)bh * N * 64;
    const unsigned short* vb = Vp + (size_t)bh * N * 64;

    // q fragments (A layout): bf16 pairs already packed in memory
    FragAB qf0, qf1;
    {
        int rowq = r0 + (lane & 15);
#pragma unroll
        for (int p = 0; p < 8; ++p) {
            int kk = a_kk(lane, p);
            qf0.u[p] = *(const unsigned*)&qb[(size_t)rowq * 64 + kk];
            qf1.u[p] = *(const unsigned*)&qb[(size_t)rowq * 64 + 32 + kk];
        }
    }

    v8f o0 = {}, o1 = {}, o2 = {}, o3 = {};
    float mstat[8], lstat[8];
#pragma unroll
    for (int r = 0; r < 8; ++r) { mstat[r] = -__builtin_inff(); lstat[r] = 0.0f; }

    unsigned short* myPs = Ps + wave * 16 * PSTR;

    for (int c0 = 0; c0 < N; c0 += 64) {
        // ---- stage K^T (manual transpose, bf16) ----
#pragma unroll
        for (int i = 0; i < 8; ++i) {
            int c = tid + i * 256;
            int t = c >> 5, d2 = (c & 31) * 2;
            unsigned u = *(const unsigned*)&kb[(size_t)(c0 + t) * 64 + d2];
            KsT[(d2 + 0) * KTSTR + t] = (unsigned short)u;
            KsT[(d2 + 1) * KTSTR + t] = (unsigned short)(u >> 16);
        }
        // ---- stage V (async DMA global->LDS, ASYNCcnt) ----
#pragma unroll
        for (int i = 0; i < 2; ++i) {
            int c = tid + i * 256;
            int t = c >> 3, d8 = (c & 7) * 8;
#ifdef HAVE_ASYNC
            __builtin_amdgcn_global_load_async_to_lds_b128(
                GPTR(&vb[(size_t)(c0 + t) * 64 + d8]),
                LPTR(&Vs[t * VSTR + d8]), 0, 0);
#else
            *(uint4*)&Vs[t * VSTR + d8] =
                *(const uint4*)&vb[(size_t)(c0 + t) * 64 + d8];
#endif
        }
        if (c0 + 64 < N) {                 // global_prefetch next k tile
            int t = tid >> 2, d = (tid & 3) * 16;
            __builtin_prefetch(&kb[(size_t)(c0 + 64 + t) * 64 + d], 0, 1);
        }
#ifdef HAVE_ASYNC
        wait_async0();
#endif
        __syncthreads();

        // ---- scores s = q @ k^T (16x64), 4 n-tiles x 2 k-steps ----
        v8f s[4];
#pragma unroll
        for (int nt = 0; nt < 4; ++nt) {
            v8f sacc = {};
#pragma unroll
            for (int kf = 0; kf < 2; ++kf) {
                FragAB bf;   // lane%16 = token column, dim pairs per a_kk
#pragma unroll
                for (int p = 0; p < 8; ++p)
                    bf.u[p] = (unsigned)KsT[(kf * 32 + a_kk(lane, p)) * KTSTR + nt * 16 + (lane & 15)] |
                              ((unsigned)KsT[(kf * 32 + a_kk(lane, p) + 1) * KTSTR + nt * 16 + (lane & 15)] << 16);
                sacc = __builtin_amdgcn_wmma_f32_16x16x32_bf16(
                    false, (kf == 0 ? qf0.v : qf1.v), false, bf.v,
                    (short)0, sacc, false, false);
            }
            s[nt] = sacc;
        }

        // ---- online softmax (DPP ring reductions within 16-lane halves) ----
        float newm[8];
#pragma unroll
        for (int r = 0; r < 8; ++r) {
            float mx = red_max16(fmaxf(fmaxf(s[0][r], s[1][r]),
                                       fmaxf(s[2][r], s[3][r])));
            float nm = fmaxf(mstat[r], mx);
            float alpha = __expf(mstat[r] - nm);
            newm[r] = nm;
            mstat[r] = nm;
            lstat[r] *= alpha;
            o0[r] *= alpha; o1[r] *= alpha; o2[r] *= alpha; o3[r] *= alpha;
        }
#pragma unroll
        for (int r = 0; r < 8; ++r) {
            float acc = 0.0f;
#pragma unroll
            for (int nt = 0; nt < 4; ++nt) {
                float pv = __expf(s[nt][r] - newm[r]);
                s[nt][r] = pv;
                acc += pv;
            }
            lstat[r] += red_sum16(acc);
        }

        // ---- relayout P: C layout -> A layout via per-wave LDS ----
#pragma unroll
        for (int nt = 0; nt < 4; ++nt)
#pragma unroll
            for (int r = 0; r < 8; ++r) {
                int m = r + 8 * (lane >> 4);
                int n = nt * 16 + (lane & 15);
                myPs[m * PSTR + n] = bf16u(s[nt][r]);
            }
        asm volatile("s_wait_dscnt 0x0" ::: "memory");  // same-wave DS order

        FragAB pf0, pf1;
        {
            int row = lane & 15;
#pragma unroll
            for (int p = 0; p < 8; ++p) {
                int kk = a_kk(lane, p);
                pf0.u[p] = *(const unsigned*)&myPs[row * PSTR + kk];
                pf1.u[p] = *(const unsigned*)&myPs[row * PSTR + 32 + kk];
            }
        }

        // ---- out += P @ V  (16x64), 4 d-tiles x 2 k-steps (tokens) ----
#pragma unroll
        for (int dt = 0; dt < 4; ++dt) {
            v8f oacc = (dt == 0) ? o0 : (dt == 1) ? o1 : (dt == 2) ? o2 : o3;
#pragma unroll
            for (int kf = 0; kf < 2; ++kf) {
                FragAB vf;  // lane%16 = dim column, token pairs per a_kk
#pragma unroll
                for (int p = 0; p < 8; ++p)
                    vf.u[p] = (unsigned)Vs[(kf * 32 + a_kk(lane, p)) * VSTR + dt * 16 + (lane & 15)] |
                              ((unsigned)Vs[(kf * 32 + a_kk(lane, p) + 1) * VSTR + dt * 16 + (lane & 15)] << 16);
                oacc = __builtin_amdgcn_wmma_f32_16x16x32_bf16(
                    false, (kf == 0 ? pf0.v : pf1.v), false, vf.v,
                    (short)0, oacc, false, false);
            }
            if (dt == 0) o0 = oacc; else if (dt == 1) o1 = oacc;
            else if (dt == 2) o2 = oacc; else o3 = oacc;
        }
        __syncthreads();
    }

    // ---- finalize: divide by row sums, write (B, N, C) ----
#pragma unroll
    for (int r = 0; r < 8; ++r) {
        float inv = (lstat[r] > 0.0f) ? (1.0f / lstat[r]) : 0.0f;
        int m = r0 + r + 8 * (lane >> 4);
        int n = lane & 15;
        size_t rowoff = ((size_t)b * 1024 + m) * 1024 + h * 64;
        O[rowoff + 0 + n]  = o0[r] * inv;
        O[rowoff + 16 + n] = o1[r] * inv;
        O[rowoff + 32 + n] = o2[r] * inv;
        O[rowoff + 48 + n] = o3[r] * inv;
    }
}

// ---------------------------------------------------------------------------
// Host-side orchestration (all on `stream`, no allocation, deterministic)
// ---------------------------------------------------------------------------
extern "C" void kernel_launch(void* const* d_in, const int* in_sizes, int n_in,
                              void* d_out, int out_size, void* d_ws, size_t ws_size,
                              hipStream_t stream) {
    (void)in_sizes; (void)n_in; (void)out_size; (void)ws_size;
    const float* x      = (const float*)d_in[0];   // (8,1024,1024)
    const float* Wqkv   = (const float*)d_in[1];   // (3072,1024)
    const float* Wproj  = (const float*)d_in[2];   // (1024,1024)
    const float* bproj  = (const float*)d_in[3];   // (1024,)
    const float* coef_k = (const float*)d_in[4];   // (10,3000)
    const float* coef_v = (const float*)d_in[5];   // (10,3000)
    const int*   indices= (const int*)d_in[6];     // (10,3000)
    const int*   taskp  = (const int*)d_in[7];     // scalar
    float* out = (float*)d_out;

    const int C = 1024, N = 1024, Btok = 8 * 1024;   // tokens = 8192
    const size_t M1 = 1024 * 1024;
    float* ws = (float*)d_ws;
    float* Bm = ws + 0 * M1;                              // 1M f32
    float* S  = ws + 1 * M1;                              // 1M f32
    float* U  = ws + 2 * M1;                              // 1M f32
    float* Wl = ws + 3 * M1;                              // 1M f32
    unsigned short* Qp = (unsigned short*)(ws + 4 * M1);  // 8M bf16
    unsigned short* Kp = (unsigned short*)(ws + 8 * M1);  // 8M bf16
    unsigned short* Vp = (unsigned short*)(ws + 12 * M1); // 8M bf16
    float* AO = ws + 16 * M1;                             // 8M f32 -> 96 MB

    dim3 blk(256);

    // 1) DCT basis
    build_dct_kernel<<<dim3((C * C + 255) / 256), blk, 0, stream>>>(Bm, C);

    // 2) qkv projection, scattered directly into packed bf16 q/k/v
    gemm_bf16_kernel<0, 1, 1><<<dim3(3 * C / GBN, Btok / GBM), blk, 0, stream>>>(
        x, C, Wqkv, C, nullptr, 0, nullptr, Btok, 3 * C, C, Qp, Kp, Vp);

    // 3) LoRA weights: W = Bm^T @ S @ Bm ; then pack-add x @ W^T into k / v
    for (int kv = 0; kv < 2; ++kv) {
        const float* coef = kv ? coef_v : coef_k;
        unsigned short* dst = kv ? Vp : Kp;
        zero_f32_kernel<<<dim3((C * C + 255) / 256), blk, 0, stream>>>(S, C * C);
        scatter_coef_kernel<<<dim3((10 * 3000 + 255) / 256), blk, 0, stream>>>(
            coef, indices, taskp, S, 3000, 10);
        // U = Bm^T @ S
        gemm_bf16_kernel<1, 0, 0><<<dim3(C / GBN, C / GBM), blk, 0, stream>>>(
            Bm, C, S, C, U, C, nullptr, C, C, C, nullptr, nullptr, nullptr);
        // Wl = U @ Bm
        gemm_bf16_kernel<0, 0, 0><<<dim3(C / GBN, C / GBM), blk, 0, stream>>>(
            U, C, Bm, C, Wl, C, nullptr, C, C, C, nullptr, nullptr, nullptr);
        // dst += pack(x @ Wl^T)
        gemm_bf16_kernel<0, 1, 2><<<dim3(C / GBN, Btok / GBM), blk, 0, stream>>>(
            x, C, Wl, C, nullptr, 0, nullptr, Btok, C, C, dst, nullptr, nullptr);
    }

    // 4) fused flash attention over all (b,h)
    flash_attn_kernel<<<dim3(N / 128, 8 * 16), blk, 0, stream>>>(Qp, Kp, Vp, AO, N);

    // 5) output projection + bias
    gemm_bf16_kernel<0, 1, 0><<<dim3(C / GBN, Btok / GBM), blk, 0, stream>>>(
        AO, C, Wproj, C, out, C, bproj, Btok, C, C, nullptr, nullptr, nullptr);
}
